// GraphLatentDiffusion_88055419502854
// MI455X (gfx1250) — matrix operand order repaired and checked by reading, not compiled, not run on hardware
//
#include <hip/hip_runtime.h>
#include <hip/hip_bf16.h>
#include <math.h>

// ---------------- problem constants ----------------
#define NNODES 10000
#define NEDGES 160000
#define EA     (NEDGES + NNODES)   // edges + self loops
#define F_IN   768
#define HID    256
#define NHEAD  4
#define D2     1024                // HID*NHEAD
#define TSTEPS 50
#define NEG_SLOPE 0.2f

// ---------------- vector types ----------------
typedef __attribute__((ext_vector_type(16))) __bf16 v16bf;
typedef __attribute__((ext_vector_type(8)))  float  v8f;
typedef __attribute__((ext_vector_type(4)))  unsigned int u32x4;
typedef __attribute__((ext_vector_type(8)))  int    i32x8;
typedef __attribute__((ext_vector_type(4)))  int    i32x4;

__device__ __forceinline__ unsigned short f32_to_bf16(float f) {
    unsigned u = __float_as_uint(f);
    unsigned r = u + 0x7FFFu + ((u >> 16) & 1u);   // round-to-nearest-even
    return (unsigned short)(r >> 16);
}

// ---------------- diffusion schedule ----------------
__global__ void sched_kernel(float* __restrict__ sa, float* __restrict__ soma) {
    if (threadIdx.x == 0 && blockIdx.x == 0) {
        const float s = 0.008f, PIH = 1.57079632679f;
        float x0 = (0.0f + s) / (1.0f + s) * PIH;
        float prev = cosf(x0); prev *= prev;
        float acp = 1.0f;
        for (int i = 1; i <= TSTEPS; ++i) {
            float xi = ((float)i / (float)TSTEPS + s) / (1.0f + s) * PIH;
            float aci = cosf(xi); aci *= aci;
            float beta = 1.0f - aci / prev;
            beta = fminf(fmaxf(beta, 0.0f), 0.999f);
            prev = aci;
            acp *= (1.0f - beta);
            sa[i - 1]   = sqrtf(acp);
            soma[i - 1] = sqrtf(1.0f - acp);
        }
    }
}

__global__ __launch_bounds__(256)
void noise_add_kernel(const float* __restrict__ emb, const float* __restrict__ noise,
                      const int* __restrict__ t, const float* __restrict__ sa,
                      const float* __restrict__ soma, float* __restrict__ out) {
    int idx = blockIdx.x * 256 + threadIdx.x;
    if (idx >= NNODES * F_IN) return;
    int node = idx / F_IN;
    int tv = t[node];
    out[idx] = sa[tv] * emb[idx] + soma[tv] * noise[idx];
}

__global__ __launch_bounds__(256)
void fill_kernel(float* __restrict__ p, float v, int n) {
    int i = blockIdx.x * 256 + threadIdx.x;
    if (i < n) p[i] = v;
}

// fp32 -> bf16 bulk convert (activations), 4 elements/thread
__global__ __launch_bounds__(256)
void cvt_bf16_kernel(const float* __restrict__ src, unsigned short* __restrict__ dst, int n4) {
    int i = blockIdx.x * 256 + threadIdx.x;
    if (i >= n4) return;
    float4 v = ((const float4*)src)[i];
    uint2 o;
    o.x = (unsigned)f32_to_bf16(v.x) | ((unsigned)f32_to_bf16(v.y) << 16);
    o.y = (unsigned)f32_to_bf16(v.z) | ((unsigned)f32_to_bf16(v.w) << 16);
    ((uint2*)dst)[i] = o;
}

// W[K,Nc] fp32 -> Wt[Nc,K] bf16 (tiled transpose through LDS)
__global__ __launch_bounds__(256)
void transpose_bf16_kernel(const float* __restrict__ W, unsigned short* __restrict__ Wt,
                           int K, int Nc) {
    __shared__ unsigned short tile[32][33];
    int k0 = blockIdx.x * 32, n0 = blockIdx.y * 32;
    int tx = threadIdx.x & 31, ty = threadIdx.x >> 5;   // 8 rows per pass
    for (int r = ty; r < 32; r += 8)
        tile[r][tx] = f32_to_bf16(W[(size_t)(k0 + r) * Nc + n0 + tx]);
    __syncthreads();
    for (int r = ty; r < 32; r += 8)
        Wt[(size_t)(n0 + r) * K + k0 + tx] = tile[tx][r];
}

// ---------------- TDM 2D tile load: global (bf16, row pitch `pitch`) -> LDS ----------------
__device__ __forceinline__ void tdm_load_2d(const unsigned short* gptr, unsigned lds_off,
                                            int rem0, int rem1, int pitch,
                                            int tile0, int tile1) {
    unsigned long long ga = (unsigned long long)(size_t)gptr;
    u32x4 g0;
    g0[0] = 1u;                                             // count=1, user mode
    g0[1] = lds_off;                                        // LDS byte address
    g0[2] = (unsigned)(ga & 0xFFFFFFFFull);                 // global_addr[31:0]
    g0[3] = (unsigned)((ga >> 32) & 0x01FFFFFFull) | (2u << 30); // addr[56:32] | type=2
    i32x8 g1;
    g1[0] = (1 << 16);                                      // data_size=1 (2 bytes)
    g1[1] = (rem0 & 0xFFFF) << 16;                          // tensor_dim0[15:0]
    g1[2] = ((rem0 >> 16) & 0xFFFF) | ((rem1 & 0xFFFF) << 16); // dim0[31:16], dim1[15:0]
    g1[3] = ((rem1 >> 16) & 0xFFFF) | ((tile0 & 0xFFFF) << 16); // dim1[31:16], tile_dim0
    g1[4] = (tile1 & 0xFFFF);                               // tile_dim1 (tile_dim2=0)
    g1[5] = pitch;                                          // tensor_dim0_stride[31:0]
    g1[6] = 0;
    g1[7] = 0;
    i32x4 gz = {};
#if defined(__clang_major__) && (__clang_major__ >= 23)
    i32x8 gz8 = {};
    __builtin_amdgcn_tensor_load_to_lds(g0, g1, gz, gz, gz8, 0);
#else
    __builtin_amdgcn_tensor_load_to_lds(g0, g1, gz, gz, 0);
#endif
}

// ---------------- bf16 WMMA GEMM:  C[M,Nc] = A[M,K] @ Wt[Nc,K]^T + bias ----------------
// BM=64, BN=128, BK=64; 128 threads = 4 waves; wave (wr,wc) owns a 32x64 tile
// = 2x4 fragments, 2 K-steps per stage -> 16 WMMA per wave per stage.
#define BM 64
#define BN 128
#define BK 64

__global__ __launch_bounds__(128)
void gemm_bf16_wmma_kernel(const unsigned short* __restrict__ A,
                           const unsigned short* __restrict__ Wt,
                           const float* __restrict__ bias, float* __restrict__ C,
                           int M, int K, int Nc) {
    __shared__ __align__(16) unsigned short lds_a[BM * BK]; // [m][k], pitch 64
    __shared__ __align__(16) unsigned short lds_b[BN * BK]; // [n][k], pitch 64

    const int tid  = threadIdx.x;
    const int wave = tid >> 5;
    const int lane = tid & 31;
    const int m0 = blockIdx.x * BM;
    const int n0 = blockIdx.y * BN;
    const int wr = wave >> 1, wc = wave & 1;   // wave tile: rows 32*wr, cols 64*wc

    v8f acc[2][4] = {};

    // A frag (16-bit A 16x32): lanes 0-15: K 0-7 & 16-23; lanes 16-31: K 8-15 & 24-31
    const int arow = lane & 15;
    const int akb  = (lane >> 4) * 8;
    // B frag (32x16): lanes 0-15: K 0-7 & 16-23; lanes 16-31: K 8-15 & 24-31; N=lane&15
    const int bcol = lane & 15;
    const int bkb  = (lane >> 4) * 8;

    const unsigned ldsA_off = (unsigned)(size_t)(void*)lds_a;
    const unsigned ldsB_off = (unsigned)(size_t)(void*)lds_b;

    union Frag { uint4 q[2]; v16bf v; };

    for (int k0 = 0; k0 < K; k0 += BK) {
        if (tid < 32) {   // TDM ignores EXEC; issue once per workgroup (wave 0)
            tdm_load_2d(A  + (size_t)m0 * K + k0, ldsA_off, K - k0, M  - m0, K, BK, BM);
            tdm_load_2d(Wt + (size_t)n0 * K + k0, ldsB_off, K - k0, Nc - n0, K, BK, BN);
            __builtin_amdgcn_s_wait_tensorcnt(0);
        }
        __syncthreads();

        #pragma unroll
        for (int ks = 0; ks < BK; ks += 32) {
            Frag af[2], bf[4];
            #pragma unroll
            for (int i = 0; i < 2; ++i) {
                const unsigned short* base = &lds_a[(32 * wr + 16 * i + arow) * BK + ks];
                af[i].q[0] = *(const uint4*)(base + akb);
                af[i].q[1] = *(const uint4*)(base + akb + 16);
            }
            #pragma unroll
            for (int j = 0; j < 4; ++j) {
                const unsigned short* base = &lds_b[(64 * wc + 16 * j + bcol) * BK + ks + bkb];
                bf[j].q[0] = *(const uint4*)(base);
                bf[j].q[1] = *(const uint4*)(base + 16);
            }
            #pragma unroll
            for (int i = 0; i < 2; ++i)
                #pragma unroll
                for (int j = 0; j < 4; ++j)
                    acc[i][j] = __builtin_amdgcn_wmma_f32_16x16x32_bf16(
                        false, af[i].v, false, bf[j].v, (short)0, acc[i][j], false, false);
        }
        __syncthreads();
    }

    // epilogue: C layout VGPR r -> row r (lanes 0-15) / row r+8 (lanes 16-31)
    const int crow8 = (lane >> 4) * 8;
    const int ccol  = lane & 15;
    #pragma unroll
    for (int i = 0; i < 2; ++i)
        #pragma unroll
        for (int j = 0; j < 4; ++j) {
            int gn = n0 + 64 * wc + 16 * j + ccol;
            float b = bias[gn];
            #pragma unroll
            for (int r = 0; r < 8; ++r) {
                int gm = m0 + 32 * wr + 16 * i + r + crow8;
                if (gm < M) C[(size_t)gm * Nc + gn] = acc[i][j][r] + b;
            }
        }
}

// ---------------- edge pipeline ----------------
__device__ __forceinline__ void edge_src_dst(int e, const int* __restrict__ ei,
                                             int& src, int& dst) {
    if (e < NEDGES) { src = ei[e]; dst = ei[NEDGES + e]; }
    else            { src = dst = e - NEDGES; }
}

__device__ __forceinline__ void atomicMaxFloat(float* addr, float v) {
    if (v >= 0.0f) atomicMax((int*)addr, __float_as_int(v));
    else           atomicMin((unsigned int*)addr, (unsigned int)__float_as_int(v));
}

__global__ __launch_bounds__(256)
void edge_logit_kernel(const float* __restrict__ xl, const float* __restrict__ xr,
                       const float* __restrict__ att, const int* __restrict__ ei,
                       float* __restrict__ logits, float* __restrict__ maxbuf) {
    int gw = (blockIdx.x * 256 + threadIdx.x) >> 5;
    int lane = threadIdx.x & 31;
    if (gw >= EA * NHEAD) return;
    int e = gw >> 2, h = gw & 3;
    int src, dst; edge_src_dst(e, ei, src, dst);

    const float* pl = xl + (size_t)src * D2 + h * HID + lane * 8;
    const float* pr = xr + (size_t)dst * D2 + h * HID + lane * 8;
    const float* pa = att + h * HID + lane * 8;
    float s = 0.0f;
    #pragma unroll
    for (int q = 0; q < 2; ++q) {
        float4 a = *(const float4*)(pl + q * 4);
        float4 b = *(const float4*)(pr + q * 4);
        float4 w = *(const float4*)(pa + q * 4);
        float v0 = a.x + b.x, v1 = a.y + b.y, v2 = a.z + b.z, v3 = a.w + b.w;
        v0 = v0 >= 0.f ? v0 : NEG_SLOPE * v0;
        v1 = v1 >= 0.f ? v1 : NEG_SLOPE * v1;
        v2 = v2 >= 0.f ? v2 : NEG_SLOPE * v2;
        v3 = v3 >= 0.f ? v3 : NEG_SLOPE * v3;
        s += w.x * v0 + w.y * v1 + w.z * v2 + w.w * v3;
    }
    #pragma unroll
    for (int off = 16; off > 0; off >>= 1) s += __shfl_xor(s, off, 32);
    if (lane == 0) {
        logits[gw] = s;
        atomicMaxFloat(&maxbuf[dst * NHEAD + h], s);
    }
}

__global__ __launch_bounds__(256)
void edge_exp_kernel(float* __restrict__ logits, const float* __restrict__ maxbuf,
                     float* __restrict__ denom, const int* __restrict__ ei) {
    int idx = blockIdx.x * 256 + threadIdx.x;
    if (idx >= EA * NHEAD) return;
    int e = idx >> 2, h = idx & 3;
    int src, dst; edge_src_dst(e, ei, src, dst);
    (void)src;
    float a = __expf(logits[idx] - maxbuf[dst * NHEAD + h]);
    logits[idx] = a;
    atomicAdd(&denom[dst * NHEAD + h], a);
}

__global__ __launch_bounds__(256)
void edge_agg_kernel(const float* __restrict__ alpha, const float* __restrict__ denom,
                     const float* __restrict__ xl, const int* __restrict__ ei,
                     float* __restrict__ agg) {
    int gw = (blockIdx.x * 256 + threadIdx.x) >> 5;
    int lane = threadIdx.x & 31;
    if (gw >= EA * NHEAD) return;
    int e = gw >> 2, h = gw & 3;
    int src, dst; edge_src_dst(e, ei, src, dst);
    float w = alpha[gw] / denom[dst * NHEAD + h];
    const float* ps = xl + (size_t)src * D2 + h * HID + lane * 8;
    float* pd = agg + (size_t)dst * D2 + h * HID + lane * 8;
    #pragma unroll
    for (int q = 0; q < 2; ++q) {
        float4 v = *(const float4*)(ps + q * 4);
        atomicAdd(pd + q * 4 + 0, w * v.x);
        atomicAdd(pd + q * 4 + 1, w * v.y);
        atomicAdd(pd + q * 4 + 2, w * v.z);
        atomicAdd(pd + q * 4 + 3, w * v.w);
    }
}

__global__ __launch_bounds__(256)
void bias_elu_kernel(const float* __restrict__ agg, const float* __restrict__ bias,
                     float* __restrict__ out) {
    int idx = blockIdx.x * 256 + threadIdx.x;
    if (idx >= NNODES * D2) return;
    float v = agg[idx] + bias[idx & (D2 - 1)];
    out[idx] = v > 0.0f ? v : expm1f(v);
}

// ---------------- launch ----------------
extern "C" void kernel_launch(void* const* d_in, const int* in_sizes, int n_in,
                              void* d_out, int out_size, void* d_ws, size_t ws_size,
                              hipStream_t stream) {
    const float* emb   = (const float*)d_in[0];
    const float* noise = (const float*)d_in[1];
    const int*   t     = (const int*)  d_in[2];
    const int*   ei    = (const int*)  d_in[3];
    const float* Wl1 = (const float*)d_in[4],  *bl1 = (const float*)d_in[5];
    const float* Wr1 = (const float*)d_in[6],  *br1 = (const float*)d_in[7];
    const float* att1 = (const float*)d_in[8], *bias1 = (const float*)d_in[9];
    const float* Wl2 = (const float*)d_in[10], *bl2 = (const float*)d_in[11];
    const float* Wr2 = (const float*)d_in[12], *br2 = (const float*)d_in[13];
    const float* att2 = (const float*)d_in[14], *bias2 = (const float*)d_in[15];
    const float* Wout = (const float*)d_in[16], *bout = (const float*)d_in[17];
    float* out = (float*)d_out;

    char* ws = (char*)d_ws;
    size_t o = 0;
    float* sa   = (float*)(ws + o); o += 256;
    float* soma = (float*)(ws + o); o += 768;                        // -> 1024
    float* X0   = (float*)(ws + o); o += (size_t)NNODES * F_IN * 4;  // x_t fp32
    float* XL   = (float*)(ws + o); o += (size_t)NNODES * D2 * 4;
    float* XR   = (float*)(ws + o); o += (size_t)NNODES * D2 * 4;
    float* AGG  = (float*)(ws + o); o += (size_t)NNODES * D2 * 4;    // agg -> h (in place)
    float* LG   = (float*)(ws + o); o += (size_t)EA * NHEAD * 4;
    float* MX   = (float*)(ws + o); o += (size_t)NNODES * NHEAD * 4;
    float* DN   = (float*)(ws + o); o += (size_t)NNODES * NHEAD * 4;
    unsigned short* Abf = (unsigned short*)(ws + o); o += (size_t)NNODES * D2 * 2; // bf16 activations
    unsigned short* WT  = (unsigned short*)(ws + o); o += (size_t)D2 * D2 * 2;     // bf16 W^T
    (void)ws_size; (void)n_in; (void)in_sizes; (void)out_size;

    const int eh = EA * NHEAD;
    dim3 blk256(256);
    dim3 gNF((NNODES * F_IN + 255) / 256);
    dim3 gND((NNODES * D2 + 255) / 256);
    dim3 gEHt((eh + 255) / 256);
    dim3 gEHw((eh * 32 + 255) / 256);
    dim3 gNH((NNODES * NHEAD + 255) / 256);
    dim3 gCv768((NNODES * F_IN / 4 + 255) / 256);
    dim3 gCv1024((NNODES * D2 / 4 + 255) / 256);
    dim3 gT_768x1024(F_IN / 32, D2 / 32);    // W[768,1024]
    dim3 gT_1024x1024(D2 / 32, D2 / 32);     // W[1024,1024]
    dim3 gT_1024x768(D2 / 32, F_IN / 32);    // W[1024,768]
    dim3 gemm_blk(128);
    dim3 gG1((NNODES + BM - 1) / BM, D2 / BN);     // -> [N,1024]
    dim3 gGo((NNODES + BM - 1) / BM, F_IN / BN);   // -> [N,768]

    // schedule + noising + bf16 activations
    sched_kernel<<<1, 64, 0, stream>>>(sa, soma);
    noise_add_kernel<<<gNF, blk256, 0, stream>>>(emb, noise, t, sa, soma, X0);
    cvt_bf16_kernel<<<gCv768, blk256, 0, stream>>>(X0, Abf, NNODES * F_IN / 4);

    // ---- layer 1 ----
    transpose_bf16_kernel<<<gT_768x1024, blk256, 0, stream>>>(Wl1, WT, F_IN, D2);
    gemm_bf16_wmma_kernel<<<gG1, gemm_blk, 0, stream>>>(Abf, WT, bl1, XL, NNODES, F_IN, D2);
    transpose_bf16_kernel<<<gT_768x1024, blk256, 0, stream>>>(Wr1, WT, F_IN, D2);
    gemm_bf16_wmma_kernel<<<gG1, gemm_blk, 0, stream>>>(Abf, WT, br1, XR, NNODES, F_IN, D2);
    fill_kernel<<<gNH, blk256, 0, stream>>>(MX, -INFINITY, NNODES * NHEAD);
    fill_kernel<<<gNH, blk256, 0, stream>>>(DN, 0.0f, NNODES * NHEAD);
    fill_kernel<<<gND, blk256, 0, stream>>>(AGG, 0.0f, NNODES * D2);
    edge_logit_kernel<<<gEHw, blk256, 0, stream>>>(XL, XR, att1, ei, LG, MX);
    edge_exp_kernel<<<gEHt, blk256, 0, stream>>>(LG, MX, DN, ei);
    edge_agg_kernel<<<gEHw, blk256, 0, stream>>>(LG, DN, XL, ei, AGG);
    bias_elu_kernel<<<gND, blk256, 0, stream>>>(AGG, bias1, AGG);    // AGG = h1
    cvt_bf16_kernel<<<gCv1024, blk256, 0, stream>>>(AGG, Abf, NNODES * D2 / 4);

    // ---- layer 2 ----
    transpose_bf16_kernel<<<gT_1024x1024, blk256, 0, stream>>>(Wl2, WT, D2, D2);
    gemm_bf16_wmma_kernel<<<gG1, gemm_blk, 0, stream>>>(Abf, WT, bl2, XL, NNODES, D2, D2);
    transpose_bf16_kernel<<<gT_1024x1024, blk256, 0, stream>>>(Wr2, WT, D2, D2);
    gemm_bf16_wmma_kernel<<<gG1, gemm_blk, 0, stream>>>(Abf, WT, br2, XR, NNODES, D2, D2);
    fill_kernel<<<gNH, blk256, 0, stream>>>(MX, -INFINITY, NNODES * NHEAD);
    fill_kernel<<<gNH, blk256, 0, stream>>>(DN, 0.0f, NNODES * NHEAD);
    fill_kernel<<<gND, blk256, 0, stream>>>(AGG, 0.0f, NNODES * D2);
    edge_logit_kernel<<<gEHw, blk256, 0, stream>>>(XL, XR, att2, ei, LG, MX);
    edge_exp_kernel<<<gEHt, blk256, 0, stream>>>(LG, MX, DN, ei);
    edge_agg_kernel<<<gEHw, blk256, 0, stream>>>(LG, DN, XL, ei, AGG);
    bias_elu_kernel<<<gND, blk256, 0, stream>>>(AGG, bias2, AGG);    // AGG = h2
    cvt_bf16_kernel<<<gCv1024, blk256, 0, stream>>>(AGG, Abf, NNODES * D2 / 4);

    // ---- output projection ----
    transpose_bf16_kernel<<<gT_1024x768, blk256, 0, stream>>>(Wout, WT, D2, F_IN);
    gemm_bf16_wmma_kernel<<<gGo, gemm_blk, 0, stream>>>(Abf, WT, bout, out, NNODES, D2, F_IN);
}
